// VRNNClassifier_59974923321591
// MI455X (gfx1250) — compile-verified, matrix operand order checked by reading, not run on hardware
//
#include <hip/hip_runtime.h>
#include <hip/hip_bf16.h>
#include <stdint.h>

// ---------------- problem constants (from reference) ----------------
#define B_TOT   4096
#define D_IN    64
#define T_STEPS 8
#define H_DIM   128
#define K_LAT   32
#define NC_CLS  10
#define M_TICKS 14
#define H3      384     // 3*H
#define BT      16      // batch rows per workgroup
#define NTHREADS 256    // 8 wave32 waves

typedef __attribute__((ext_vector_type(16))) __bf16 v16bf;
typedef __attribute__((ext_vector_type(8)))  float  v8f;

// ---------------- math helpers ----------------
__device__ __forceinline__ float sigmf_(float x) { return 1.0f / (1.0f + __expf(-x)); }
__device__ __forceinline__ float softplusf_(float v) {
  return fmaxf(v, 0.0f) + log1pf(__expf(-fabsf(v)));
}
__device__ __forceinline__ uint32_t hash_u32(uint32_t x) {
  x ^= x >> 16; x *= 0x7feb352dU; x ^= x >> 15; x *= 0x846ca68bU; x ^= x >> 16; return x;
}
// deterministic pseudo-normal (Box-Muller on counter hash)
__device__ __forceinline__ float gauss_(uint32_t idx) {
  uint32_t a = hash_u32(idx * 2654435761U + 0x9e3779b9U);
  uint32_t b = hash_u32(a + 0x85ebca6bU);
  float u1 = ((a >> 8) + 1u) * (1.0f / 16777217.0f);   // (0,1]
  float u2 = (b >> 8) * (1.0f / 16777216.0f);
  return sqrtf(-2.0f * __logf(u1)) * __cosf(6.28318530718f * u2);
}

// ---------------- WMMA (gfx1250, wave32) ----------------
__device__ __forceinline__ v8f wmma_bf16(v16bf a, v16bf b, v8f c) {
  // v_wmma_f32_16x16x32_bf16 : D = A(16x32) * B(32x16) + C(16x16 f32)
  return __builtin_amdgcn_wmma_f32_16x16x32_bf16(false, a, false, b, (short)0, c,
                                                 false, false);
}

__device__ __forceinline__ void pack8(v16bf& f, int base, float4 a, float4 b) {
  f[base + 0] = (__bf16)a.x; f[base + 1] = (__bf16)a.y;
  f[base + 2] = (__bf16)a.z; f[base + 3] = (__bf16)a.w;
  f[base + 4] = (__bf16)b.x; f[base + 5] = (__bf16)b.y;
  f[base + 6] = (__bf16)b.z; f[base + 7] = (__bf16)b.w;
}

// A fragment (16x32, 16-bit): lane holds row = lane&15; K slots:
// kb..kb+7 and kb+16..kb+23 where kb = (lane>=16 ? 8 : 0)  (ISA 7.12.2)
// float4 LDS loads -> ds_load_b128 (kb is a multiple of 8 floats => 32B aligned)
__device__ __forceinline__ v16bf afrag_lds(const float* S, int ld, int k0) {
  int lane = threadIdx.x & 31;
  int row  = lane & 15;
  int kb   = k0 + ((lane & 16) ? 8 : 0);
  const float4* p = reinterpret_cast<const float4*>(S + row * ld + kb);
  float4 a0 = p[0], a1 = p[1];   // k = kb .. kb+7
  float4 c0 = p[4], c1 = p[5];   // k = kb+16 .. kb+23
  v16bf f;
  pack8(f, 0, a0, a1);
  pack8(f, 8, c0, c1);
  return f;
}

// B fragment (32x16): lane holds col = lane&15; same K striping as A.
// W is row-major [n_out][k_in]; B[k][n] = W[n0+n][k0+k]  (gemm vs W^T)
__device__ __forceinline__ v16bf bfrag_g(const float* W, int ld, int n0, int k0,
                                         bool active) {
  int lane = threadIdx.x & 31;
  int n    = lane & 15;
  int kb   = k0 + ((lane & 16) ? 8 : 0);
  v16bf f;
  if (active) {
    const float4* p = reinterpret_cast<const float4*>(W + (n0 + n) * ld + kb);
    float4 a0 = p[0], a1 = p[1];
    float4 c0 = p[4], c1 = p[5];
    pack8(f, 0, a0, a1);
    pack8(f, 8, c0, c1);
  } else {
#pragma unroll
    for (int q = 0; q < 16; ++q) f[q] = (__bf16)0.0f;
  }
  return f;
}

// C/D tile (16x16 f32, 8 VGPRs): lane col = lane&15, rows mb+v, mb = lane>=16 ? 8:0
__device__ __forceinline__ void store_tile_bias(float* dst, int ld, int n0, v8f acc,
                                                const float* bias) {
  int lane = threadIdx.x & 31;
  int n    = lane & 15;
  int mb   = (lane & 16) ? 8 : 0;
  float bv = bias[n0 + n];
#pragma unroll
  for (int v = 0; v < 8; ++v)
    dst[(mb + v) * ld + n0 + n] = acc[v] + bv;
}

// ---------------- LDS layout (dynamic, ~92 KB) ----------------
#define SM_S    0
#define SM_H    (SM_S   + BT * H_DIM)        // 2048
#define SM_GI   (SM_H   + BT * H_DIM)        // 2048
#define SM_GH   (SM_GI  + BT * H3)           // 6144
#define SM_XI   (SM_GH  + BT * H3)           // 6144
#define SM_ENC  (SM_XI  + BT * D_IN)         // 1024
#define SM_Z    (SM_ENC + BT * 2 * K_LAT)    // 1024
#define SM_NZ   (SM_Z   + BT * K_LAT)        // 512
#define SM_YH   (SM_NZ  + BT * K_LAT)        // 512
#define SM_ST   (SM_YH  + BT * 16)           // 256
#define SM_YT   (SM_ST  + BT * H_DIM)        // 2048
#define SM_BIH  (SM_YT  + BT * 16)           // 256
#define SM_BHH  (SM_BIH + H3)
#define SM_BE   (SM_BHH + H3)
#define SM_BL   (SM_BE  + 2 * K_LAT)
#define SM_BO   (SM_BL  + H_DIM)
#define SM_CUM  (SM_BO  + 16)
#define SM_PM   (SM_CUM + BT)
#define SM_PH   (SM_PM  + BT)
#define SM_RT   (SM_PH  + BT)
#define SM_NT   (SM_RT  + BT)
#define SM_TOTAL (SM_NT + BT)
#define SMEM_BYTES (SM_TOTAL * 4)

__global__ __launch_bounds__(NTHREADS, 1)
void vrnn_act_kernel(const float* __restrict__ x,   const float* __restrict__ sx,
                     const float* __restrict__ Wih, const float* __restrict__ Whh,
                     const float* __restrict__ bih, const float* __restrict__ bhh,
                     const float* __restrict__ We,  const float* __restrict__ be,
                     const float* __restrict__ Wl,  const float* __restrict__ bl,
                     const float* __restrict__ Wo,  const float* __restrict__ bo,
                     const float* __restrict__ Whalt, const float* __restrict__ bhalt,
                     float* __restrict__ out) {
  extern __shared__ float smem[];
  float* lds_s   = smem + SM_S;
  float* lds_h   = smem + SM_H;
  float* lds_gi  = smem + SM_GI;
  float* lds_gh  = smem + SM_GH;
  float* lds_xi  = smem + SM_XI;
  float* lds_enc = smem + SM_ENC;
  float* lds_z   = smem + SM_Z;
  float* lds_nz  = smem + SM_NZ;
  float* lds_yh  = smem + SM_YH;
  float* lds_st  = smem + SM_ST;
  float* lds_yt  = smem + SM_YT;
  float* c_bih   = smem + SM_BIH;
  float* c_bhh   = smem + SM_BHH;
  float* c_be    = smem + SM_BE;
  float* c_bl    = smem + SM_BL;
  float* c_bo    = smem + SM_BO;
  float* a_cum   = smem + SM_CUM;
  float* a_pm    = smem + SM_PM;
  float* a_ph    = smem + SM_PH;
  float* a_rt    = smem + SM_RT;
  int*   a_nt    = (int*)(smem + SM_NT);

  const int tid = threadIdx.x;
  const int w   = tid >> 5;                    // wave id, 0..7
  const int b0  = blockIdx.x * BT;             // batch tile origin

  // ---- VGPR-resident bf16 weight fragments (loaded once, reused 112 ticks) ----
  v16bf whh_fr[3][4];   // wave w owns output cols 48*w .. 48*w+47 of 3H
  v16bf wih_fr[3][2];
  v16bf we_fr[4];       // waves 0..3: cols 16*w of 2K
  v16bf wl_fr;          // wave w: cols 16*w of H
  v16bf woh_fr[4];      // wave 0: combined [Wo ; Whalt] 16-col head
#pragma unroll
  for (int c = 0; c < 3; ++c) {
    int n0 = 16 * (3 * w + c);
#pragma unroll
    for (int ks = 0; ks < 4; ++ks) whh_fr[c][ks] = bfrag_g(Whh, H_DIM, n0, 32 * ks, true);
#pragma unroll
    for (int ks = 0; ks < 2; ++ks) wih_fr[c][ks] = bfrag_g(Wih, D_IN, n0, 32 * ks, true);
  }
#pragma unroll
  for (int ks = 0; ks < 4; ++ks) we_fr[ks] = bfrag_g(We, H_DIM, 16 * w, 32 * ks, w < 4);
  wl_fr = bfrag_g(Wl, K_LAT, 16 * w, 0, true);
  {
    int lane = tid & 31;
    int c    = lane & 15;
    int kb   = (lane & 16) ? 8 : 0;
#pragma unroll
    for (int ks = 0; ks < 4; ++ks) {
      v16bf f;
#pragma unroll
      for (int q = 0; q < 16; ++q) {
        int k   = 32 * ks + kb + (q < 8 ? q : 8 + q);
        float v = 0.0f;
        if (w == 0) {
          if (c < NC_CLS)       v = Wo[c * H_DIM + k];
          else if (c == NC_CLS) v = Whalt[k];
        }
        f[q] = (__bf16)v;
      }
      woh_fr[ks] = f;
    }
  }

  // ---- init persistent LDS state ----
  for (int i = tid; i < BT * H_DIM / 4; i += NTHREADS) {
    reinterpret_cast<float4*>(lds_s)[i] =
        reinterpret_cast<const float4*>(sx + (size_t)b0 * H_DIM)[i];
    reinterpret_cast<float4*>(lds_st)[i] = make_float4(0.f, 0.f, 0.f, 0.f);
  }
  for (int i = tid; i < H3; i += NTHREADS) { c_bih[i] = bih[i]; c_bhh[i] = bhh[i]; }
  for (int i = tid; i < 2 * K_LAT; i += NTHREADS) c_be[i] = be[i];
  for (int i = tid; i < H_DIM; i += NTHREADS) c_bl[i] = bl[i];
  if (tid < 16) c_bo[tid] = (tid < NC_CLS) ? bo[tid] : 0.0f;
  if (tid < BT) { a_cum[tid] = 0.f; a_pm[tid] = 0.f; a_ph[tid] = 0.f; a_rt[tid] = 0.f; a_nt[tid] = -1; }
  if (tid < BT * 16 / 4)
    reinterpret_cast<float4*>(lds_yt)[tid] = make_float4(0.f, 0.f, 0.f, 0.f);
  const float bhalt0 = bhalt[0];
  __syncthreads();

  float* outY = out;
  float* outP = out + (size_t)B_TOT * NC_CLS * T_STEPS;
  float* outN = outP + (size_t)B_TOT * T_STEPS;
  float* outS = outN + (size_t)B_TOT * T_STEPS;

#pragma unroll 1
  for (int t = 0; t < T_STEPS; ++t) {
    // stage x_t tile [BT x D]  (x layout: (B, D, T); stride T in d -> scalar gathers)
    for (int i = tid; i < BT * D_IN; i += NTHREADS) {
      int b = i >> 6, d = i & 63;
      lds_xi[i] = x[((size_t)(b0 + b) * D_IN + d) * T_STEPS + t];
    }
    if (t + 1 < T_STEPS)  // CDNA5 global_prefetch of next timestep slice
      __builtin_prefetch(&x[((size_t)(b0 + (tid >> 4)) * D_IN + ((tid & 15) << 2)) * T_STEPS + t + 1], 0, 1);
    __syncthreads();

    // gi = x_t @ Wih^T + bih   (tick-invariant; hoisted out of M loop)
    {
      v16bf aX[2];
#pragma unroll
      for (int ks = 0; ks < 2; ++ks) aX[ks] = afrag_lds(lds_xi, D_IN, 32 * ks);
#pragma unroll
      for (int c = 0; c < 3; ++c) {
        int n0 = 16 * (3 * w + c);
        v8f acc = {0.f, 0.f, 0.f, 0.f, 0.f, 0.f, 0.f, 0.f};
#pragma unroll
        for (int ks = 0; ks < 2; ++ks) acc = wmma_bf16(aX[ks], wih_fr[c][ks], acc);
        store_tile_bias(lds_gi, H3, n0, acc, c_bih);
      }
    }
    __syncthreads();

#pragma unroll 1
    for (int m = 0; m < M_TICKS; ++m) {
      // gh = s @ Whh^T + bhh  : [16x384], 24 tiles, 3 per wave, K=128 (4 wmma steps)
      {
        v16bf aS[4];
#pragma unroll
        for (int ks = 0; ks < 4; ++ks) aS[ks] = afrag_lds(lds_s, H_DIM, 32 * ks);
#pragma unroll
        for (int c = 0; c < 3; ++c) {
          int n0 = 16 * (3 * w + c);
          v8f acc = {0.f, 0.f, 0.f, 0.f, 0.f, 0.f, 0.f, 0.f};
#pragma unroll
          for (int ks = 0; ks < 4; ++ks) acc = wmma_bf16(aS[ks], whh_fr[c][ks], acc);
          store_tile_bias(lds_gh, H3, n0, acc, c_bhh);
        }
      }
      __syncthreads();

      // GRU elementwise -> h (float4 lanes: 512 vec4 elements, 2 per thread)
      for (int i = tid; i < BT * H_DIM / 4; i += NTHREADS) {
        int b  = i >> 5;              // 32 vec4 per row of H=128
        int j4 = (i & 31) << 2;
        const float* gi = lds_gi + b * H3;
        const float* gh = lds_gh + b * H3;
        float4 gir = *reinterpret_cast<const float4*>(gi + j4);
        float4 giz = *reinterpret_cast<const float4*>(gi + H_DIM + j4);
        float4 gin = *reinterpret_cast<const float4*>(gi + 2 * H_DIM + j4);
        float4 ghr = *reinterpret_cast<const float4*>(gh + j4);
        float4 ghz = *reinterpret_cast<const float4*>(gh + H_DIM + j4);
        float4 ghn = *reinterpret_cast<const float4*>(gh + 2 * H_DIM + j4);
        float4 sv  = reinterpret_cast<const float4*>(lds_s)[i];
        float4 hv;
        {
          float r = sigmf_(gir.x + ghr.x), zg = sigmf_(giz.x + ghz.x);
          float nn = tanhf(gin.x + r * ghn.x);
          hv.x = (1.0f - zg) * nn + zg * sv.x;
        }
        {
          float r = sigmf_(gir.y + ghr.y), zg = sigmf_(giz.y + ghz.y);
          float nn = tanhf(gin.y + r * ghn.y);
          hv.y = (1.0f - zg) * nn + zg * sv.y;
        }
        {
          float r = sigmf_(gir.z + ghr.z), zg = sigmf_(giz.z + ghz.z);
          float nn = tanhf(gin.z + r * ghn.z);
          hv.z = (1.0f - zg) * nn + zg * sv.z;
        }
        {
          float r = sigmf_(gir.w + ghr.w), zg = sigmf_(giz.w + ghz.w);
          float nn = tanhf(gin.w + r * ghn.w);
          hv.w = (1.0f - zg) * nn + zg * sv.w;
        }
        reinterpret_cast<float4*>(lds_h)[i] = hv;
      }
      __syncthreads();

      // enc = h @ We^T + be (waves 0-3)  ||  gaussian noise (waves 4-7)
      if (w < 4) {
        v8f acc = {0.f, 0.f, 0.f, 0.f, 0.f, 0.f, 0.f, 0.f};
#pragma unroll
        for (int ks = 0; ks < 4; ++ks) {
          v16bf a = afrag_lds(lds_h, H_DIM, 32 * ks);
          acc = wmma_bf16(a, we_fr[ks], acc);
        }
        store_tile_bias(lds_enc, 2 * K_LAT, 16 * w, acc, c_be);
      } else {
        int i = tid - 128;                       // 128 threads, 1 vec4 each
        if (i < BT * K_LAT / 4) {
          int b = i >> 3, kk4 = (i & 7) << 2;
          uint32_t base = (uint32_t)(((t * M_TICKS + m) * B_TOT + b0 + b) * K_LAT + kk4);
          float4 nz = make_float4(gauss_(base), gauss_(base + 1),
                                  gauss_(base + 2), gauss_(base + 3));
          reinterpret_cast<float4*>(lds_nz)[i] = nz;
        }
      }
      __syncthreads();

      // z = noise * softplus(enc[:,K:]-5) + enc[:,:K]  (128 vec4)
      if (tid < BT * K_LAT / 4) {
        int i = tid;
        int b = i >> 3, kk4 = (i & 7) << 2;
        const float* e = lds_enc + b * 2 * K_LAT;
        float4 mu = *reinterpret_cast<const float4*>(e + kk4);
        float4 sg = *reinterpret_cast<const float4*>(e + K_LAT + kk4);
        float4 nz = reinterpret_cast<const float4*>(lds_nz)[i];
        float4 zv;
        zv.x = nz.x * softplusf_(sg.x - 5.0f) + mu.x;
        zv.y = nz.y * softplusf_(sg.y - 5.0f) + mu.y;
        zv.z = nz.z * softplusf_(sg.z - 5.0f) + mu.z;
        zv.w = nz.w * softplusf_(sg.w - 5.0f) + mu.w;
        reinterpret_cast<float4*>(lds_z)[i] = zv;
      }
      __syncthreads();

      // hb = tanh(z @ Wl^T + bl) -> becomes new state s (8 tiles, 1 per wave)
      {
        v16bf a  = afrag_lds(lds_z, K_LAT, 0);
        v8f acc  = {0.f, 0.f, 0.f, 0.f, 0.f, 0.f, 0.f, 0.f};
        acc = wmma_bf16(a, wl_fr, acc);
        int lane = tid & 31;
        int n = lane & 15, mb = (lane & 16) ? 8 : 0, n0 = 16 * w;
        float bv = c_bl[n0 + n];
#pragma unroll
        for (int v = 0; v < 8; ++v)
          lds_s[(mb + v) * H_DIM + n0 + n] = tanhf(acc[v] + bv);
      }
      __syncthreads();

      // y / pn head: one 16-col tile combining Wo rows 0..9 + Whalt at col 10 (wave 0)
      if (w == 0) {
        v8f acc = {0.f, 0.f, 0.f, 0.f, 0.f, 0.f, 0.f, 0.f};
#pragma unroll
        for (int ks = 0; ks < 4; ++ks) {
          v16bf a = afrag_lds(lds_s, H_DIM, 32 * ks);
          acc = wmma_bf16(a, woh_fr[ks], acc);
        }
        int lane = tid & 31;
        int c = lane & 15, mb = (lane & 16) ? 8 : 0;
#pragma unroll
        for (int v = 0; v < 8; ++v) {
          float val;
          if (c < NC_CLS)       val = acc[v] + c_bo[c];
          else if (c == NC_CLS) val = sigmf_(acc[v] + bhalt0);
          else                  val = 0.0f;
          lds_yh[(mb + v) * 16 + c] = val;
        }
      }
      __syncthreads();

      // ACT bookkeeping per batch row
      if (tid < BT) {
        int b    = tid;
        float pn = lds_yh[b * 16 + NC_CLS];
        float raw = a_cum[b] + pn;
        float pm  = fminf(1.0f, raw);
        if (m == M_TICKS - 1) pm = 1.0f;
        float ph = pm - a_pm[b];
        if (pm >= 1.0f && a_nt[b] < 0) {
          a_nt[b] = m;
          a_rt[b] = (m == 0) ? 0.0f : 1.0f - a_pm[b];
        }
        a_cum[b] = raw;
        a_pm[b]  = pm;
        a_ph[b]  = ph;
      }
      __syncthreads();

      // weighted accumulation of state / logits (vec4)
      for (int i = tid; i < BT * H_DIM / 4; i += NTHREADS) {
        float p = a_ph[i >> 5];
        float4 sv = reinterpret_cast<const float4*>(lds_s)[i];
        float4 st = reinterpret_cast<float4*>(lds_st)[i];
        st.x += p * sv.x; st.y += p * sv.y; st.z += p * sv.z; st.w += p * sv.w;
        reinterpret_cast<float4*>(lds_st)[i] = st;
      }
      if (tid < BT * NC_CLS) {
        int b = tid / NC_CLS, c = tid % NC_CLS;
        lds_yt[b * 16 + c] += a_ph[b] * lds_yh[b * 16 + c];
      }
      __syncthreads();
    } // m

    // emit per-timestep outputs; roll st -> s ; reset accumulators
    if (tid < BT * NC_CLS) {
      int b = tid / NC_CLS, c = tid % NC_CLS;
      outY[((size_t)(b0 + b) * NC_CLS + c) * T_STEPS + t] = lds_yt[b * 16 + c];
    }
    if (tid < BT) {
      float ntf = (float)a_nt[tid];
      outP[(size_t)(b0 + tid) * T_STEPS + t] = ntf + a_rt[tid];
      outN[(size_t)(b0 + tid) * T_STEPS + t] = ntf;
    }
    __syncthreads();
    for (int i = tid; i < BT * H_DIM / 4; i += NTHREADS) {
      float4 v = reinterpret_cast<const float4*>(lds_st)[i];
      reinterpret_cast<float4*>(lds_s)[i] = v;
      if (t == T_STEPS - 1)
        reinterpret_cast<float4*>(outS + (size_t)b0 * H_DIM)[i] = v;
      reinterpret_cast<float4*>(lds_st)[i] = make_float4(0.f, 0.f, 0.f, 0.f);
    }
    if (tid < BT) { a_cum[tid] = 0.f; a_pm[tid] = 0.f; a_ph[tid] = 0.f; a_rt[tid] = 0.f; a_nt[tid] = -1; }
    if (tid < BT * 16 / 4)
      reinterpret_cast<float4*>(lds_yt)[tid] = make_float4(0.f, 0.f, 0.f, 0.f);
    __syncthreads();
  } // t
}

extern "C" void kernel_launch(void* const* d_in, const int* in_sizes, int n_in,
                              void* d_out, int out_size, void* d_ws, size_t ws_size,
                              hipStream_t stream) {
  (void)in_sizes; (void)n_in; (void)out_size; (void)d_ws; (void)ws_size;
  const float* x     = (const float*)d_in[0];
  const float* sx    = (const float*)d_in[1];
  const float* Wih   = (const float*)d_in[2];
  const float* Whh   = (const float*)d_in[3];
  const float* bih   = (const float*)d_in[4];
  const float* bhh   = (const float*)d_in[5];
  const float* We    = (const float*)d_in[6];
  const float* be    = (const float*)d_in[7];
  const float* Wl    = (const float*)d_in[8];
  const float* bl    = (const float*)d_in[9];
  const float* Wo    = (const float*)d_in[10];
  const float* bo    = (const float*)d_in[11];
  const float* Whalt = (const float*)d_in[12];
  const float* bhalt = (const float*)d_in[13];
  float* out = (float*)d_out;

  (void)hipFuncSetAttribute(reinterpret_cast<const void*>(vrnn_act_kernel),
                            hipFuncAttributeMaxDynamicSharedMemorySize,
                            (int)SMEM_BYTES);
  dim3 grid(B_TOT / BT), block(NTHREADS);
  vrnn_act_kernel<<<grid, block, SMEM_BYTES, stream>>>(
      x, sx, Wih, Whh, bih, bhh, We, be, Wl, bl, Wo, bo, Whalt, bhalt, out);
}
